// RelationAwareAttention_41369124995874
// MI455X (gfx1250) — matrix operand order repaired and checked by reading, not compiled, not run on hardware
//
#include <hip/hip_runtime.h>

#define DIM   1024
#define NHEAD 16
#define HDIM  64

typedef __bf16 bf16;
typedef __attribute__((ext_vector_type(16))) __bf16 v16bf;
typedef __attribute__((ext_vector_type(8)))  __bf16 v8bf;
typedef __attribute__((ext_vector_type(4)))  __bf16 v4bf;
typedef __attribute__((ext_vector_type(8)))  float  v8f;
typedef __attribute__((ext_vector_type(4)))  float  v4f;

// ---------------------------------------------------------------------------
// WMMA helpers (CDNA5 16x16x32 bf16, f32 accumulate)
// ---------------------------------------------------------------------------
__device__ __forceinline__ v8f wmma_bf16(v16bf a, v16bf b, v8f c) {
  // (neg_a, A, neg_b, B, c_mod, C, reuse_a, reuse_b)
  return __builtin_amdgcn_wmma_f32_16x16x32_bf16(false, a, false, b, (short)0, c,
                                                 false, false);
}

// A-matrix 16x32 fragment: lane (l&15)=row, h=l>>4.
// elements 0..15 = row[16h+0..15]
__device__ __forceinline__ v16bf load_afrag(const bf16* rowp, int half) {
  v8bf lo = *(const v8bf*)(rowp + 16 * half);
  v8bf hi = *(const v8bf*)(rowp + 16 * half + 8);
  v16bf r;
#pragma unroll
  for (int i = 0; i < 8; ++i) { r[i] = lo[i]; r[i + 8] = hi[i]; }
  return r;
}

// B-matrix 32x16 fragment from Bt (row-major N x K): lane col = l&15,
// elements = Bt[col][kOff + 0..15], kOff = 16*(l>>4). 32B contiguous load.
__device__ __forceinline__ v16bf load_bfrag(const bf16* p) {
  return *(const v16bf*)p;
}

// ---------------------------------------------------------------------------
// Prep: fp32 -> bf16 (vectorized x4)
// ---------------------------------------------------------------------------
__global__ void f32_to_bf16_kernel(const float* __restrict__ in,
                                   bf16* __restrict__ out, int n4) {
  int i = blockIdx.x * blockDim.x + threadIdx.x;
  if (i < n4) {
    v4f v = ((const v4f*)in)[i];
    v4bf o;
#pragma unroll
    for (int j = 0; j < 4; ++j) o[j] = (bf16)v[j];
    ((v4bf*)out)[i] = o;
  }
}

// ---------------------------------------------------------------------------
// Prep: Wt[n][k] = (bf16) W[k][n]   (DIM x DIM), 32x32 LDS tiles
// ---------------------------------------------------------------------------
__global__ __launch_bounds__(256) void transpose_to_bf16_kernel(
    const float* __restrict__ W, bf16* __restrict__ Wt) {
  __shared__ float tile[32][33];
  int tx = threadIdx.x & 31, ty = threadIdx.x >> 5;
  int nb = blockIdx.x * 32, kb = blockIdx.y * 32;
#pragma unroll
  for (int i = 0; i < 4; ++i)
    tile[ty + 8 * i][tx] = W[(size_t)(kb + ty + 8 * i) * DIM + nb + tx];
  __syncthreads();
#pragma unroll
  for (int i = 0; i < 4; ++i)
    Wt[(size_t)(nb + ty + 8 * i) * DIM + kb + tx] = (bf16)tile[tx][ty + 8 * i];
}

// ---------------------------------------------------------------------------
// GEMM:  C[M x DIM] = A[M x DIM](bf16) * Bt^T + bias,  M % 16 == 0.
// One wave -> 16x64 output strip. K-loop: 128-elem macro-steps, one prefetch
// per stream per macro-step, fully unrolled 4x(1 A frag + 4 B frags + 4 WMMA).
// ---------------------------------------------------------------------------
template <bool F32OUT>
__global__ __launch_bounds__(256) void gemm_kernel(
    const bf16* __restrict__ A, const bf16* __restrict__ Bt,
    const float* __restrict__ bias, void* __restrict__ Cout, int M) {
  const int K = DIM;
  int lane = threadIdx.x & 31;
  int wave = blockIdx.x * (blockDim.x >> 5) + (threadIdx.x >> 5);
  int mTiles = M >> 4;
  int nStrips = DIM >> 6;  // 16
  if (wave >= mTiles * nStrips) return;
  int mt = wave >> 4;       // waves in a block share mt -> A reuse in cache
  int nt = wave & (nStrips - 1);
  int half = lane >> 4, col = lane & 15;

  const bf16* arow = A + (size_t)(mt * 16 + col) * K + 16 * half;
  const bf16* brow[4];
#pragma unroll
  for (int t = 0; t < 4; ++t)
    brow[t] = Bt + (size_t)(nt * 64 + t * 16 + col) * K + 16 * half;

  v8f acc[4] = {v8f{}, v8f{}, v8f{}, v8f{}};
  for (int kb = 0; kb < K; kb += 128) {
    // global_prefetch_b8: keep the A/B streams warm in GL2 (256B/stream ahead)
    if (kb + 128 < K) {
      __builtin_prefetch(arow + kb + 128, 0, 3);
#pragma unroll
      for (int t = 0; t < 4; ++t) __builtin_prefetch(brow[t] + kb + 128, 0, 3);
    }
#pragma unroll
    for (int ku = 0; ku < 128; ku += 32) {
      int kk = kb + ku;
      v8bf alo = *(const v8bf*)(arow + kk);
      v8bf ahi = *(const v8bf*)(arow + kk + 8);
      v16bf a;
#pragma unroll
      for (int i = 0; i < 8; ++i) { a[i] = alo[i]; a[i + 8] = ahi[i]; }
#pragma unroll
      for (int t = 0; t < 4; ++t)
        acc[t] = wmma_bf16(a, load_bfrag(brow[t] + kk), acc[t]);
    }
  }

#pragma unroll
  for (int t = 0; t < 4; ++t) {
    int coln = nt * 64 + t * 16 + col;
    float bv = bias[coln];
#pragma unroll
    for (int r = 0; r < 8; ++r) {
      int row = mt * 16 + r + 8 * half;
      float v = acc[t][r] + bv;
      if (F32OUT)
        ((float*)Cout)[(size_t)row * DIM + coln] = v;
      else
        ((bf16*)Cout)[(size_t)row * DIM + coln] = (bf16)v;
    }
  }
}

// ---------------------------------------------------------------------------
// Flash attention per (image, head). Block = 256 thr = 8 waves.
// Wave owns a 16-row Q tile; streams keys 32/step.
//   S^T = K * Q^T         (C-layout: lane column == q row -> per-lane softmax)
//   O^T = V^T * P^T       (V^T staged transposed in LDS)
// K tile staged with CDNA5 async global->LDS DMA (ASYNCcnt).
// ---------------------------------------------------------------------------
__global__ __launch_bounds__(256) void attn_kernel(
    const bf16* __restrict__ q, const bf16* __restrict__ k,
    const bf16* __restrict__ v, const int* __restrict__ lens,
    bf16* __restrict__ o) {
  // single static LDS object -> allocated at LDS offset 0
  __shared__ __align__(16) char smem[65536];
  bf16* Klds = (bf16*)smem;            // [key][d]  256x64  32 KB (offset 0)
  bf16* Vt = (bf16*)(smem + 32768);    // [d][key]  64x256  32 KB

  const int head = blockIdx.x;
  const int img = blockIdx.y;
  const float scale = 1.0f / 32.0f;  // 1/sqrt(DIM)

  int n0 = 0;
  for (int j = 0; j < img; ++j) n0 += lens[j];
  const int L = lens[img];
  const int Lpad32 = (L + 31) & ~31;
  const int L8 = L * 8, Lp8 = Lpad32 * 8;

  // ---- stage K: async global -> LDS, 16B per lane per op (keys < L) ----
  {
    const bf16* kbase = k + (size_t)n0 * DIM + head * HDIM;
    for (int c = threadIdx.x; c < L8; c += blockDim.x) {
      int key = c >> 3, part = c & 7;
      unsigned voff = (unsigned)(key * 2048 + part * 16);   // global byte off
      unsigned ldsoff = (unsigned)(key * 128 + part * 16);  // LDS byte off
      asm volatile("global_load_async_to_lds_b128 %0, %1, %2"
                   :
                   : "v"(ldsoff), "v"(voff), "s"(kbase)
                   : "memory");
    }
    // zero-fill padded keys [L, Lpad32) with normal LDS stores (disjoint)
    for (int c = threadIdx.x; c < Lp8 - L8; c += blockDim.x) {
      int cc = c + L8;
      v8bf z = {};
      *(v8bf*)&Klds[(cc >> 3) * 64 + (cc & 7) * 8] = z;
    }
  }

  // ---- stage V transposed: 16B vector loads, scalar LDS scatter ----
  {
    const bf16* vbase = v + (size_t)n0 * DIM + head * HDIM;
    for (int c = threadIdx.x; c < Lp8; c += blockDim.x) {
      int key = c >> 3, part = c & 7;
      v8bf vv = {};
      if (key < L) vv = *(const v8bf*)(vbase + (size_t)key * DIM + part * 8);
#pragma unroll
      for (int j = 0; j < 8; ++j) Vt[(part * 8 + j) * 256 + key] = vv[j];
    }
  }

  asm volatile("s_wait_asynccnt 0" ::: "memory");
  __syncthreads();

  const int wid = threadIdx.x >> 5, lane = threadIdx.x & 31;
  const int half = lane >> 4, col = lane & 15;
  const int nQT = (L + 15) >> 4;

  for (int qt = wid; qt < nQT; qt += 8) {
    int qrow = qt * 16 + col;
    int qld = qrow < L ? qrow : L - 1;  // clamp for safe loads
    const bf16* qp = q + (size_t)(n0 + qld) * DIM + head * HDIM;
    v16bf qb0 = load_bfrag(qp + half * 16);        // d 0..31
    v16bf qb1 = load_bfrag(qp + 32 + half * 16);   // d 32..63

    v8f ot[4] = {v8f{}, v8f{}, v8f{}, v8f{}};
    float m = -1e30f, l = 0.0f;

    for (int kb = 0; kb < Lpad32; kb += 32) {
      const bf16* krow0 = &Klds[(kb + col) * 64];
      const bf16* krow1 = &Klds[(kb + 16 + col) * 64];
      v8f s0 = {}, s1 = {};
      s0 = wmma_bf16(load_afrag(krow0, half), qb0, s0);
      s0 = wmma_bf16(load_afrag(krow0 + 32, half), qb1, s0);
      s1 = wmma_bf16(load_afrag(krow1, half), qb0, s1);
      s1 = wmma_bf16(load_afrag(krow1 + 32, half), qb1, s1);

      // s0[r] = S^T[key=kb+8h+r][q=col], s1 shifted by 16 keys
      float sv0[8], sv1[8];
      float tm = -1e30f;
#pragma unroll
      for (int r = 0; r < 8; ++r) {
        int key0 = kb + 8 * half + r;
        sv0[r] = (key0 < L) ? s0[r] * scale : -1e30f;
        sv1[r] = (key0 + 16 < L) ? s1[r] * scale : -1e30f;
        tm = fmaxf(tm, fmaxf(sv0[r], sv1[r]));
      }
      tm = fmaxf(tm, __shfl_xor(tm, 16, 32));
      float mn = fmaxf(m, tm);
      float alpha = __expf(m - mn);
      float rs = 0.0f;
#pragma unroll
      for (int r = 0; r < 8; ++r) {
        sv0[r] = __expf(sv0[r] - mn);
        sv1[r] = __expf(sv1[r] - mn);
        rs += sv0[r] + sv1[r];
      }
      rs += __shfl_xor(rs, 16, 32);
      l = l * alpha + rs;
      m = mn;
#pragma unroll
      for (int t = 0; t < 4; ++t)
#pragma unroll
        for (int r = 0; r < 8; ++r) ot[t][r] *= alpha;

      // assemble P^T B-fragment: lane needs 16 consecutive keys of its q col
      v16bf pb;
#pragma unroll
      for (int r = 0; r < 8; ++r) {
        float o0 = __shfl_xor(sv0[r], 16, 32);
        float o1 = __shfl_xor(sv1[r], 16, 32);
        pb[r]     = (bf16)(half ? o1 : sv0[r]);  // keys kOff+0..7
        pb[r + 8] = (bf16)(half ? sv1[r] : o0);  // keys kOff+8..15
      }

      // O^T += V^T * P^T  (4 d-tiles of 16)
#pragma unroll
      for (int t = 0; t < 4; ++t) {
        const bf16* vrow = &Vt[(t * 16 + col) * 256 + kb];
        ot[t] = wmma_bf16(load_afrag(vrow, half), pb, ot[t]);
      }
    }

    if (qrow < L) {
      float linv = (l > 0.0f) ? 1.0f / l : 0.0f;
      bf16* op = o + (size_t)(n0 + qrow) * DIM + head * HDIM;
#pragma unroll
      for (int t = 0; t < 4; ++t) {
        v8bf pk;
#pragma unroll
        for (int r = 0; r < 8; ++r) pk[r] = (bf16)(ot[t][r] * linv);
        *(v8bf*)(op + t * 16 + 8 * half) = pk;  // d = t*16 + 8h + r
      }
    }
  }
}

// ---------------------------------------------------------------------------
// host launch
// ---------------------------------------------------------------------------
extern "C" void kernel_launch(void* const* d_in, const int* in_sizes, int n_in,
                              void* d_out, int out_size, void* d_ws,
                              size_t ws_size, hipStream_t stream) {
  const float* x  = (const float*)d_in[0];
  const float* Wq = (const float*)d_in[1];
  const float* bq = (const float*)d_in[2];
  const float* Wk = (const float*)d_in[3];
  const float* bk = (const float*)d_in[4];
  const float* Wv = (const float*)d_in[5];
  const float* bv = (const float*)d_in[6];
  const float* Wu = (const float*)d_in[7];
  const float* bu = (const float*)d_in[8];
  const int* lens = (const int*)d_in[9];
  const int Ntok = in_sizes[0] / DIM;  // 12224
  const int B = in_sizes[9];           // 64

  size_t off = 0;
  auto alloc = [&](size_t bytes) {
    size_t o = off;
    off += (bytes + 255) & ~(size_t)255;
    return o;
  };
  char* ws = (char*)d_ws;
  const size_t tokBytes = (size_t)Ntok * DIM * sizeof(bf16);
  const size_t wBytes = (size_t)DIM * DIM * sizeof(bf16);
  bf16* xb   = (bf16*)(ws + alloc(tokBytes));
  bf16* qb   = (bf16*)(ws + alloc(tokBytes));
  bf16* kb   = (bf16*)(ws + alloc(tokBytes));
  bf16* vb   = (bf16*)(ws + alloc(tokBytes));
  bf16* ob   = (bf16*)(ws + alloc(tokBytes));
  bf16* WqT  = (bf16*)(ws + alloc(wBytes));
  bf16* WkT  = (bf16*)(ws + alloc(wBytes));
  bf16* WvT  = (bf16*)(ws + alloc(wBytes));
  bf16* WuT  = (bf16*)(ws + alloc(wBytes));
  (void)ws_size;

  int n4 = Ntok * DIM / 4;
  f32_to_bf16_kernel<<<(n4 + 255) / 256, 256, 0, stream>>>(x, xb, n4);

  dim3 tg(DIM / 32, DIM / 32);
  transpose_to_bf16_kernel<<<tg, 256, 0, stream>>>(Wq, WqT);
  transpose_to_bf16_kernel<<<tg, 256, 0, stream>>>(Wk, WkT);
  transpose_to_bf16_kernel<<<tg, 256, 0, stream>>>(Wv, WvT);
  transpose_to_bf16_kernel<<<tg, 256, 0, stream>>>(Wu, WuT);

  int jobs = (Ntok >> 4) * (DIM >> 6);
  int gblocks = (jobs + 7) / 8;
  gemm_kernel<false><<<gblocks, 256, 0, stream>>>(xb, WqT, bq, qb, Ntok);
  gemm_kernel<false><<<gblocks, 256, 0, stream>>>(xb, WkT, bk, kb, Ntok);
  gemm_kernel<false><<<gblocks, 256, 0, stream>>>(xb, WvT, bv, vb, Ntok);

  attn_kernel<<<dim3(NHEAD, B), 256, 0, stream>>>(qb, kb, vb, lens, ob);

  gemm_kernel<true><<<gblocks, 256, 0, stream>>>(ob, WuT, bu, d_out, Ntok);
}
